// ContextualAttention_61735859913439
// MI455X (gfx1250) — matrix operand order repaired and checked by reading, not compiled, not run on hardware
//
#include <hip/hip_runtime.h>
#include <hip/hip_bf16.h>
#include <math.h>

// ---------------------------------------------------------------------------
// ContextualAttention for MI455X (gfx1250, wave32, WMMA)
//   X:    (4,128,64,64) f32      mask: (4,1,256,256) f32
//   out:  (4,128,64,64) f32
// Pipeline:
//   pack_patches -> Pm (B,1024,1152) f16 + norms
//   pack_deconv  -> Dq (B,2048,1024) f16  (q-major)
//   compute_mm   -> mm (1024) f32
//   wmma_gemm_nt -> G = Pm * Pm^T  (B,1024,1024) f32
//   softmax_yi   -> yit (B,1024p,1024l) f16  (transposed attention)
//   wmma_gemm_nt -> T = Dq * yit^T (B,2048,1024) f32
//   overlap_add  -> out
// GEMM: one wave owns a 32x64 output tile (2 A-frags x 4 B-frags -> 8 v_wmma
// per K-step). All 4 B fragments are loaded into distinct registers before the
// WMMA block so loadcnt can drain incrementally instead of waiting to zero.
// ---------------------------------------------------------------------------

typedef __attribute__((ext_vector_type(16))) _Float16 v16h;
typedef __attribute__((ext_vector_type(8)))  _Float16 v8h;
typedef __attribute__((ext_vector_type(8)))  float    v8f;

__device__ __forceinline__ v16h join8(v8h a, v8h b) {
  return __builtin_shufflevector(a, b, 0, 1, 2, 3, 4, 5, 6, 7,
                                 8, 9, 10, 11, 12, 13, 14, 15);
}

// ---------------------------------------------------------------------------
// Pack 3x3 patches of Xd = X[:, :, ::2, ::2] into Pm[b*1024+l][c*9+u*3+v] (f16)
// and compute f32 L2 norms per patch. One block per (b,l); 128 thr = channels.
// ---------------------------------------------------------------------------
__global__ void pack_patches(const float* __restrict__ X,
                             _Float16* __restrict__ Pm,
                             float* __restrict__ norms) {
  int bl = blockIdx.x;            // b*1024 + l
  int b  = bl >> 10;
  int l  = bl & 1023;
  int i  = l >> 5, j = l & 31;    // patch center in 32x32
  int c  = threadIdx.x;           // channel 0..127

  const float* Xb = X + (size_t)(b * 128 + c) * 4096;  // 64*64 plane
  _Float16* row = Pm + (size_t)bl * 1152 + c * 9;

  float ssq = 0.f;
#pragma unroll
  for (int u = 0; u < 3; ++u)
#pragma unroll
    for (int v = 0; v < 3; ++v) {
      int y = i + u - 1, x = j + v - 1;
      float val = (y >= 0 && y < 32 && x >= 0 && x < 32)
                      ? Xb[(2 * y) * 64 + 2 * x] : 0.f;
      row[u * 3 + v] = (_Float16)val;
      ssq += val * val;
    }

  __shared__ float red[128];
  red[c] = ssq;
  __syncthreads();
  for (int s = 64; s > 0; s >>= 1) {
    if (c < s) red[c] += red[c + s];
    __syncthreads();
  }
  if (c == 0) norms[bl] = sqrtf(red[0]);
}

// ---------------------------------------------------------------------------
// Pack 4x4 stride-2 "deconv" patches of full-res X, q-major:
// Dq[b][q=(c,u,v)][l=(i,j)] = X[b][c][2i+u-1][2j+v-1] (0 OOB), f16.
// ---------------------------------------------------------------------------
__global__ void pack_deconv(const float* __restrict__ X,
                            _Float16* __restrict__ Dq) {
  long idx = (long)blockIdx.x * 256 + threadIdx.x;   // < 4*2048*1024
  int l = (int)(idx & 1023);
  int q = (int)((idx >> 10) & 2047);
  int b = (int)(idx >> 21);
  int c = q >> 4, u = (q >> 2) & 3, v = q & 3;
  int i = l >> 5, j = l & 31;
  int y = 2 * i + u - 1, x = 2 * j + v - 1;
  float val = (y >= 0 && y < 64 && x >= 0 && x < 64)
                  ? X[(size_t)(b * 128 + c) * 4096 + y * 64 + x] : 0.f;
  Dq[idx] = (_Float16)val;
}

// ---------------------------------------------------------------------------
// mm[l] = 1 iff the 3x3 patch of mask[0,0,::8,::8] around l is all zero.
// ---------------------------------------------------------------------------
__global__ void compute_mm(const float* __restrict__ mask,
                           float* __restrict__ mm) {
  int l = blockIdx.x * 256 + threadIdx.x;   // 0..1023
  int i = l >> 5, j = l & 31;
  float any = 0.f;
#pragma unroll
  for (int u = 0; u < 3; ++u)
#pragma unroll
    for (int v = 0; v < 3; ++v) {
      int y = i + u - 1, x = j + v - 1;
      if (y >= 0 && y < 32 && x >= 0 && x < 32)
        any += fabsf(mask[(8 * y) * 256 + 8 * x]);
    }
  mm[l] = (any == 0.f) ? 1.f : 0.f;
}

// ---------------------------------------------------------------------------
// Generic batched WMMA GEMM, C = A * B^T  (both A and B row-major, K fastest):
//   C[b][m][n] = sum_k A[b][m][k] * B[b][n][k]        (f16 in, f32 out)
// One wave per 32x64 output tile: 2 A fragments x 4 B fragments, 8 f32
// accumulators, 8 v_wmma per K-step. 256 threads = 8 waves per block.
// Exact grid: waves = batches * mtiles32 * nquads (no divergence -> EXEC all-1s
// around every WMMA, as required).
// ---------------------------------------------------------------------------
__global__ void wmma_gemm_nt(const _Float16* __restrict__ A,
                             const _Float16* __restrict__ B,
                             float* __restrict__ C,
                             int lda, int ldb, int ldc,
                             int ksteps, int mtiles32, int nquads,
                             size_t sA, size_t sB, size_t sC) {
  int wave    = blockIdx.x * (blockDim.x >> 5) + (threadIdx.x >> 5);
  int lane    = threadIdx.x & 31;
  int lane_lo = lane & 15;
  int hi      = lane >> 4;

  int per_batch = mtiles32 * nquads;
  int b   = wave / per_batch;
  int rem = wave - b * per_batch;
  int mt  = rem / nquads;
  int nq  = rem - mt * nquads;

  // A fragment rows: lane_lo = M; lanes 0-15 hold K 0..15, lanes 16-31 hold
  // K 16..31, contiguous per lane (matching B).
  const _Float16* Arow0 = A + b * sA + (size_t)(mt * 32 + lane_lo) * lda;
  const _Float16* Arow1 = Arow0 + (size_t)16 * lda;
  // B fragment: lane_lo = N (a row of B since C = A*B^T); lanes 0-15 hold
  // K 0..15, lanes 16-31 hold K 16..31, contiguous per lane.
  const _Float16* Brow0 = B + b * sB + (size_t)(nq * 64 + lane_lo) * ldb;
  const _Float16* Brow1 = Brow0 + (size_t)16 * ldb;
  const _Float16* Brow2 = Brow0 + (size_t)32 * ldb;
  const _Float16* Brow3 = Brow0 + (size_t)48 * ldb;

  v8f acc[2][4] = {};

  for (int ks = 0; ks < ksteps; ++ks) {
    int k0 = ks * 32;
    // Issue all operand loads for this K-step up front (distinct registers),
    // so the scheduler can clause them and drain loadcnt incrementally while
    // the first WMMAs execute.
    v16h a0 = join8(*(const v8h*)(Arow0 + k0 + hi * 16),
                    *(const v8h*)(Arow0 + k0 + hi * 16 + 8));
    v16h a1 = join8(*(const v8h*)(Arow1 + k0 + hi * 16),
                    *(const v8h*)(Arow1 + k0 + hi * 16 + 8));
    v16h bf0 = join8(*(const v8h*)(Brow0 + k0 + hi * 16),
                     *(const v8h*)(Brow0 + k0 + hi * 16 + 8));
    v16h bf1 = join8(*(const v8h*)(Brow1 + k0 + hi * 16),
                     *(const v8h*)(Brow1 + k0 + hi * 16 + 8));
    v16h bf2 = join8(*(const v8h*)(Brow2 + k0 + hi * 16),
                     *(const v8h*)(Brow2 + k0 + hi * 16 + 8));
    v16h bf3 = join8(*(const v8h*)(Brow3 + k0 + hi * 16),
                     *(const v8h*)(Brow3 + k0 + hi * 16 + 8));

    // Stream-ahead prefetch into the near (WGP) cache level; speculative,
    // safely dropped past the end of the buffer.
    __builtin_prefetch(Arow0 + k0 + 64, 0, 3);
    __builtin_prefetch(Arow1 + k0 + 64, 0, 3);
    __builtin_prefetch(Brow0 + k0 + 64, 0, 3);
    __builtin_prefetch(Brow2 + k0 + 64, 0, 3);

    acc[0][0] = __builtin_amdgcn_wmma_f32_16x16x32_f16(
        false, a0, false, bf0, (short)0, acc[0][0], false, false);
    acc[1][0] = __builtin_amdgcn_wmma_f32_16x16x32_f16(
        false, a1, false, bf0, (short)0, acc[1][0], false, false);
    acc[0][1] = __builtin_amdgcn_wmma_f32_16x16x32_f16(
        false, a0, false, bf1, (short)0, acc[0][1], false, false);
    acc[1][1] = __builtin_amdgcn_wmma_f32_16x16x32_f16(
        false, a1, false, bf1, (short)0, acc[1][1], false, false);
    acc[0][2] = __builtin_amdgcn_wmma_f32_16x16x32_f16(
        false, a0, false, bf2, (short)0, acc[0][2], false, false);
    acc[1][2] = __builtin_amdgcn_wmma_f32_16x16x32_f16(
        false, a1, false, bf2, (short)0, acc[1][2], false, false);
    acc[0][3] = __builtin_amdgcn_wmma_f32_16x16x32_f16(
        false, a0, false, bf3, (short)0, acc[0][3], false, false);
    acc[1][3] = __builtin_amdgcn_wmma_f32_16x16x32_f16(
        false, a1, false, bf3, (short)0, acc[1][3], false, false);
  }

  // D layout: VGPR r -> (M = r + 8*hi, N = lane_lo)
#pragma unroll
  for (int m = 0; m < 2; ++m) {
    float* Crow = C + b * sC +
                  (size_t)(mt * 32 + m * 16 + hi * 8) * ldc + nq * 64 + lane_lo;
#pragma unroll
    for (int t = 0; t < 4; ++t)
#pragma unroll
      for (int r = 0; r < 8; ++r)
        Crow[(size_t)r * ldc + t * 16] = acc[m][t][r];
  }
}

// ---------------------------------------------------------------------------
// Masked softmax over l for each (b,p); writes transposed attention yit[b][p][l]
// (f16) so GEMM2's B operand is K(=l)-contiguous.
// z_l = SCALE * mm_l * G[b][p][l] / max(norm_l, EPS); masked entries enter the
// softmax as 0 (matching reference), then output is multiplied by mm_l.
// ---------------------------------------------------------------------------
__global__ void softmax_yi(const float* __restrict__ G,
                           const float* __restrict__ norms,
                           const float* __restrict__ mm,
                           _Float16* __restrict__ yit) {
  int bp  = blockIdx.x;     // b*1024 + p
  int b   = bp >> 10;
  int tid = threadIdx.x;

  const float* Grow = G + (size_t)bp * 1024;
  const float* nb   = norms + b * 1024;

  __shared__ float red[256];

  float z[4];
  float mloc = -1e30f;
#pragma unroll
  for (int r = 0; r < 4; ++r) {
    int l = r * 256 + tid;
    float zz = 10.0f * mm[l] * Grow[l] / fmaxf(nb[l], 1e-4f);
    z[r] = zz;
    mloc = fmaxf(mloc, zz);
  }
  red[tid] = mloc;
  __syncthreads();
  for (int s = 128; s > 0; s >>= 1) {
    if (tid < s) red[tid] = fmaxf(red[tid], red[tid + s]);
    __syncthreads();
  }
  float zmax = red[0];
  __syncthreads();

  float sum = 0.f;
#pragma unroll
  for (int r = 0; r < 4; ++r) {
    z[r] = __expf(z[r] - zmax);
    sum += z[r];
  }
  red[tid] = sum;
  __syncthreads();
  for (int s = 128; s > 0; s >>= 1) {
    if (tid < s) red[tid] += red[tid + s];
    __syncthreads();
  }
  float inv = 1.0f / red[0];

  _Float16* yrow = yit + (size_t)bp * 1024;
#pragma unroll
  for (int r = 0; r < 4; ++r) {
    int l = r * 256 + tid;
    yrow[l] = (_Float16)(mm[l] * z[r] * inv);
  }
}

// ---------------------------------------------------------------------------
// Overlap-add gather: out[b][c][y][x] = 0.25 * sum over valid (u,v) taps of
// T[b][c*16+u*4+v][i*32+j] with 2i+u = y+1, 2j+v = x+1.
// ---------------------------------------------------------------------------
__global__ void overlap_add(const float* __restrict__ T,
                            float* __restrict__ out) {
  int idx = blockIdx.x * 256 + threadIdx.x;  // < 4*128*64*64
  int x = idx & 63;
  int y = (idx >> 6) & 63;
  int c = (idx >> 12) & 127;
  int b = idx >> 19;

  const float* Tb = T + (size_t)b * 2048 * 1024;
  int u0 = (y + 1) & 1, v0 = (x + 1) & 1;
  float acc = 0.f;
#pragma unroll
  for (int du = 0; du < 2; ++du) {
    int u = u0 + 2 * du;
    int i = (y + 1 - u) >> 1;
    if (i < 0 || i > 31) continue;
#pragma unroll
    for (int dv = 0; dv < 2; ++dv) {
      int v = v0 + 2 * dv;
      int j = (x + 1 - v) >> 1;
      if (j < 0 || j > 31) continue;
      acc += Tb[(size_t)(c * 16 + u * 4 + v) * 1024 + i * 32 + j];
    }
  }
  out[idx] = acc * 0.25f;
}

// ---------------------------------------------------------------------------
extern "C" void kernel_launch(void* const* d_in, const int* in_sizes, int n_in,
                              void* d_out, int out_size, void* d_ws, size_t ws_size,
                              hipStream_t stream) {
  const float* X    = (const float*)d_in[0];  // 4*128*64*64
  const float* mask = (const float*)d_in[1];  // 4*1*256*256
  float* out        = (float*)d_out;          // 4*128*64*64

  char* ws = (char*)d_ws;
  size_t off = 0;
  _Float16* Pm   = (_Float16*)(ws + off); off += (size_t)4 * 1024 * 1152 * 2; // 9,437,184
  float*    norms= (float*)   (ws + off); off += (size_t)4 * 1024 * 4;        // 16,384
  float*    mm   = (float*)   (ws + off); off += (size_t)1024 * 4;            // 4,096
  float*    G    = (float*)   (ws + off); off += (size_t)4 * 1024 * 1024 * 4; // 16,777,216
  _Float16* Dq   = (_Float16*)(ws + off); off += (size_t)4 * 2048 * 1024 * 2; // 16,777,216
  _Float16* yit  = (_Float16*)(ws + off); off += (size_t)4 * 1024 * 1024 * 2; // 8,388,608
  float*    T    = (float*)   (ws + off); off += (size_t)4 * 2048 * 1024 * 4; // 33,554,432
  // total workspace: 84,955,136 bytes

  // 1) patch matrix + norms
  pack_patches<<<4 * 1024, 128, 0, stream>>>(X, Pm, norms);
  // 2) deconv patch matrix (q-major)
  pack_deconv<<<(4 * 2048 * 1024) / 256, 256, 0, stream>>>(X, Dq);
  // 3) background-patch mask
  compute_mm<<<4, 256, 0, stream>>>(mask, mm);
  // 4) GEMM1: G = Pm * Pm^T   (M=1024, N=1024, K=1152 per batch)
  //    waves = 4 * (1024/32) * (1024/64) = 2048 -> 256 blocks of 8 waves
  wmma_gemm_nt<<<256, 256, 0, stream>>>(Pm, Pm, G,
                                        1152, 1152, 1024,
                                        36, 32, 16,
                                        (size_t)1024 * 1152,
                                        (size_t)1024 * 1152,
                                        (size_t)1024 * 1024);
  // 5) masked softmax -> transposed attention (f16)
  softmax_yi<<<4 * 1024, 256, 0, stream>>>(G, norms, mm, yit);
  // 6) GEMM2: T = Dq * yit^T  (M=2048, N=1024, K=1024 per batch)
  //    waves = 4 * (2048/32) * (1024/64) = 4096 -> 512 blocks
  wmma_gemm_nt<<<512, 256, 0, stream>>>(Dq, yit, T,
                                        1024, 1024, 1024,
                                        32, 64, 16,
                                        (size_t)2048 * 1024,
                                        (size_t)1024 * 1024,
                                        (size_t)2048 * 1024);
  // 7) overlap-add to output
  overlap_add<<<(4 * 128 * 64 * 64) / 256, 256, 0, stream>>>(T, out);
}
